// SLMDecoderBlock_5497558139366
// MI455X (gfx1250) — compile-verified, matrix operand order checked by reading, not run on hardware
//
#include <hip/hip_runtime.h>

// ---------------------------------------------------------------------------
// CDNA5 (gfx1250, wave32) decoder block: RMSNorm -> QKV -> RoPE -> causal GQA
// flash attention -> O proj (+res) -> RMSNorm -> SwiGLU MLP (+res).
// All matmuls via v_wmma_f32_16x16x32_bf16 (f32 accum). Weights pre-transposed
// to [N,K] so A- and B-fragment staging are both contiguous 16B copies.
// GEMM K-loop double-buffers LDS tiles with gfx1250 async global->LDS loads;
// all LDS fragment loads are hoisted ahead of dense WMMA bursts.
// ---------------------------------------------------------------------------

typedef __bf16 bf16;
typedef __attribute__((ext_vector_type(16))) __bf16 v16bf;
typedef __attribute__((ext_vector_type(4)))  __bf16 v4bf;
typedef __attribute__((ext_vector_type(8)))  float   v8f;
typedef __attribute__((ext_vector_type(4)))  unsigned int u32x4;

#define HID   2048
#define NH    32
#define NKV   8
#define HD    64
#define FF    5632
#define BB    2
#define SS    2048
#define TOK   (BB * SS)
#define EPSV  1e-6f
#define THETA 10000.0f

union Frag { v16bf v; u32x4 q[2]; };

__device__ __forceinline__ void load_frag(Frag& f, const bf16* p) {
  f.q[0] = *reinterpret_cast<const u32x4*>(p);
  f.q[1] = *reinterpret_cast<const u32x4*>(p + 16);
}

// ---------------------------------------------------------------------------
// Async global->LDS 16B copy (ASYNCcnt-tracked on gfx1250), sync fallback.
// Builtin params are pointers to vector_size(16) int in AS1 (global) / AS3
// (LDS) — cast to exactly those types.
// ---------------------------------------------------------------------------
#if __has_builtin(__builtin_amdgcn_global_load_async_to_lds_b128)
#define HAS_ASYNC_LDS 1
#else
#define HAS_ASYNC_LDS 0
#endif

typedef int v4i_gcc __attribute__((vector_size(16)));
typedef __attribute__((address_space(1))) v4i_gcc* as1_v4i;
typedef __attribute__((address_space(3))) v4i_gcc* as3_v4i;

__device__ __forceinline__ void async_cp16(bf16* lds_dst, const bf16* gsrc) {
#if HAS_ASYNC_LDS
  __builtin_amdgcn_global_load_async_to_lds_b128(
      (as1_v4i)gsrc, (as3_v4i)lds_dst, 0, 0);
#else
  *reinterpret_cast<u32x4*>(lds_dst) = *reinterpret_cast<const u32x4*>(gsrc);
#endif
}

__device__ __forceinline__ void async_wait0() {
#if HAS_ASYNC_LDS
#if __has_builtin(__builtin_amdgcn_s_wait_asynccnt)
  __builtin_amdgcn_s_wait_asynccnt(0);
#else
  asm volatile("s_wait_asynccnt 0" ::: "memory");
#endif
#endif
}

// ---------------------------------------------------------------------------
// Tiled transpose + convert: W[K,N] fp32 -> Wt[N,K] bf16
// ---------------------------------------------------------------------------
__global__ __launch_bounds__(256) void transpose_bf16_kernel(
    const float* __restrict__ in, bf16* __restrict__ out, int K, int N) {
  __shared__ bf16 tile[32][33];
  const int kb = blockIdx.y * 32, nb = blockIdx.x * 32;
  const int tx = threadIdx.x & 31, ty = threadIdx.x >> 5;   // 32 x 8
#pragma unroll
  for (int r = ty; r < 32; r += 8)
    tile[r][tx] = (bf16)in[(size_t)(kb + r) * N + nb + tx];
  __syncthreads();
#pragma unroll
  for (int r = ty; r < 32; r += 8)
    out[(size_t)(nb + r) * K + kb + tx] = tile[tx][r];
}

// ---------------------------------------------------------------------------
// RMSNorm over H=2048, one 256-thread block per row, bf16 output
// ---------------------------------------------------------------------------
__global__ __launch_bounds__(256) void rmsnorm_bf16_kernel(
    const float* __restrict__ x, const float* __restrict__ w, bf16* __restrict__ out) {
  __shared__ float red[8];
  const int row = blockIdx.x;
  const float* xr = x + (size_t)row * HID;
  float vals[8];
  float ss = 0.f;
#pragma unroll
  for (int j = 0; j < 8; ++j) {
    float v = xr[threadIdx.x + j * 256];
    vals[j] = v;
    ss += v * v;
  }
#pragma unroll
  for (int m = 16; m >= 1; m >>= 1) ss += __shfl_xor(ss, m, 32);
  if ((threadIdx.x & 31) == 0) red[threadIdx.x >> 5] = ss;
  __syncthreads();
  float tot = 0.f;
#pragma unroll
  for (int wv = 0; wv < 8; ++wv) tot += red[wv];
  const float rstd = rsqrtf(tot * (1.0f / (float)HID) + EPSV);
#pragma unroll
  for (int j = 0; j < 8; ++j) {
    int c = threadIdx.x + j * 256;
    out[(size_t)row * HID + c] = (bf16)(vals[j] * rstd * w[c]);
  }
}

// ---------------------------------------------------------------------------
// RoPE + relayout: in [B,S,nheads*HD] fp32 -> out [B,nheads,S,HD] bf16
// (or transposed [B,nheads,HD,S] when vt=1, used for V so the attention
//  kernel can stage V^T tiles with contiguous 16B copies).
// ---------------------------------------------------------------------------
__global__ void rope_relayout_kernel(const float* __restrict__ in, bf16* __restrict__ out,
                                     int nheads, int do_rope, int vt) {
  int i = blockIdx.x * blockDim.x + threadIdx.x;
  int d = i & 31;
  int rest = i >> 5;
  int s = rest & (SS - 1);
  rest >>= 11;                      // SS == 2^11
  int h = rest % nheads;
  int b = rest / nheads;
  if (b >= BB) return;
  size_t inb = ((size_t)(b * SS + s) * nheads + h) * HD;
  float x1 = in[inb + d], x2 = in[inb + d + 32];
  float o1, o2;
  if (do_rope) {
    float inv = __powf(THETA, -(float)(2 * d) * (1.0f / (float)HD));
    float ang = (float)s * inv;
    float sn, cs;
    __sincosf(ang, &sn, &cs);
    o1 = x1 * cs - x2 * sn;
    o2 = x2 * cs + x1 * sn;
  } else {
    o1 = x1; o2 = x2;
  }
  if (vt) {
    size_t ob = ((size_t)(b * nheads + h) * HD + d) * SS + s;
    out[ob]           = (bf16)o1;
    out[ob + 32 * SS] = (bf16)o2;
  } else {
    size_t ob = ((size_t)(b * nheads + h) * SS + s) * HD;
    out[ob + d]      = (bf16)o1;
    out[ob + d + 32] = (bf16)o2;
  }
}

// ---------------------------------------------------------------------------
// bf16 GEMM: C[M,N] = A[M,K] * Bt[N,K]^T   (f32 accum)
// Block tile 128x128, K-step 32, 8 waves (2x4), each wave 64x32 (4x2 WMMA).
// Double-buffered LDS tiles streamed with async global->LDS b128 copies.
// All six fragments are loaded before the 8-WMMA burst so the loads pipeline
// under one DScnt wait and the matrix pipe runs back-to-back.
// Epilogue: RES adds fp32 residual; OUTBF16 writes bf16 instead of f32.
// ---------------------------------------------------------------------------
template <int RES, int OUTBF16>
__global__ __launch_bounds__(256) void gemm_bf16_kernel(
    const bf16* __restrict__ A, const bf16* __restrict__ Bt,
    float* __restrict__ Cf, bf16* __restrict__ Cb, const float* __restrict__ Res,
    int M, int N, int K) {
  constexpr int LDT = 40;                 // 32 + 8 pad; 80B rows, 16B aligned
  __shared__ bf16 As[2][128 * LDT];
  __shared__ bf16 Bs[2][128 * LDT];

  const int tid  = threadIdx.x;
  const int lane = tid & 31, wave = tid >> 5;
  const int l16  = lane & 15, half = (lane >> 4) & 1;
  const int wm   = wave >> 2, wn = wave & 3;
  const int m0   = blockIdx.y * 128, n0 = blockIdx.x * 128;

  // staging coords: 512 x 16B chunks per tile, 2 per thread per tile
  const int r0  = tid >> 2;               // 0..63
  const int off = (tid & 3) * 8;          // 0,8,16,24

  auto stage = [&](int buf, int k0) {
    async_cp16(&As[buf][(r0     ) * LDT + off], A  + (size_t)(m0 + r0     ) * K + k0 + off);
    async_cp16(&As[buf][(r0 + 64) * LDT + off], A  + (size_t)(m0 + r0 + 64) * K + k0 + off);
    async_cp16(&Bs[buf][(r0     ) * LDT + off], Bt + (size_t)(n0 + r0     ) * K + k0 + off);
    async_cp16(&Bs[buf][(r0 + 64) * LDT + off], Bt + (size_t)(n0 + r0 + 64) * K + k0 + off);
  };

  v8f acc[4][2] = {};
  const int nk = K >> 5;
  stage(0, 0);
  for (int it = 0; it < nk; ++it) {
    async_wait0();
    __syncthreads();                       // tiles for `it` visible to all
    if (it + 1 < nk) stage((it + 1) & 1, (it + 1) << 5);
    const bf16* as = As[it & 1];
    const bf16* bs = Bs[it & 1];
    // hoist ALL fragment loads ahead of the WMMA burst
    Frag bfr[2], af[4];
#pragma unroll
    for (int j = 0; j < 2; ++j)
      load_frag(bfr[j], &bs[(wn * 32 + j * 16 + l16) * LDT + half * 8]);
#pragma unroll
    for (int i = 0; i < 4; ++i)
      load_frag(af[i], &as[(wm * 64 + i * 16 + l16) * LDT + half * 8]);
#pragma unroll
    for (int i = 0; i < 4; ++i)
#pragma unroll
      for (int j = 0; j < 2; ++j)
        acc[i][j] = __builtin_amdgcn_wmma_f32_16x16x32_bf16(
            false, af[i].v, false, bfr[j].v, (short)0, acc[i][j], false, false);
  }

#pragma unroll
  for (int i = 0; i < 4; ++i)
#pragma unroll
    for (int j = 0; j < 2; ++j)
#pragma unroll
      for (int p = 0; p < 8; ++p) {
        int row = m0 + wm * 64 + i * 16 + p + half * 8;
        int col = n0 + wn * 32 + j * 16 + l16;
        float v = acc[i][j][p];
        if (RES) v += Res[(size_t)row * N + col];
        if (OUTBF16) Cb[(size_t)row * N + col] = (bf16)v;
        else         Cf[(size_t)row * N + col] = v;
      }
}

// ---------------------------------------------------------------------------
// Causal GQA flash attention. Q [B,NH,S,HD], K [B,NKV,S,HD], Vt [B,NKV,HD,S].
// Out: [B,S,NH*HD] bf16. 4 waves x 16 query rows; 32 keys per LDS pass.
// ---------------------------------------------------------------------------
__global__ __launch_bounds__(128) void attn_kernel(
    const bf16* __restrict__ Q, const bf16* __restrict__ Kc,
    const bf16* __restrict__ Vt, bf16* __restrict__ O) {
  constexpr int LDK = 72, LDV = 40, LDP = 40;
  __shared__ bf16 Ks [32 * LDK];
  __shared__ bf16 Vts[64 * LDV];
  __shared__ bf16 Ps [4 * 16 * LDP];

  const int tid  = threadIdx.x;
  const int lane = tid & 31, wave = tid >> 5;
  const int l16  = lane & 15, half = (lane >> 4) & 1;
  const int q0   = blockIdx.x * 64;
  const int h    = blockIdx.y;
  const int b    = blockIdx.z;
  const int kvh  = h / (NH / NKV);

  const bf16* qbase  = Q  + ((size_t)(b * NH  + h  ) * SS) * HD;
  const bf16* kbase  = Kc + ((size_t)(b * NKV + kvh) * SS) * HD;
  const bf16* vtbase = Vt + ((size_t)(b * NKV + kvh) * HD) * SS;

  // Q fragments (rows q0 + wave*16 + l16), HD=64 -> 2 fragments over d
  Frag qf[2];
  {
    const bf16* qr = qbase + (size_t)(q0 + wave * 16 + l16) * HD;
#pragma unroll
    for (int ks = 0; ks < 2; ++ks)
      load_frag(qf[ks], qr + ks * 32 + half * 8);
  }

  float mrow[8], lrow[8];
#pragma unroll
  for (int p = 0; p < 8; ++p) { mrow[p] = -3.0e38f; lrow[p] = 0.f; }
  v8f Oacc[4] = {};

  const int nkt = (q0 >> 5) + 2;           // key tiles covering [0, q0+64)
  for (int kt = 0; kt < nkt; ++kt) {
    const int k0 = kt * 32;
    __syncthreads();                       // previous pass finished reading
    // K tile row-major (32 x 64), V^T tile (64 d-rows x 32 keys): 16B copies
#pragma unroll
    for (int c = tid; c < 256; c += 128) {
      int kk = c >> 3, dd = (c & 7) * 8;
      async_cp16(&Ks[kk * LDK + dd], kbase + (size_t)(k0 + kk) * HD + dd);
    }
#pragma unroll
    for (int c = tid; c < 256; c += 128) {
      int dd = c >> 2, ko = (c & 3) * 8;
      async_cp16(&Vts[dd * LDV + ko], vtbase + (size_t)dd * SS + k0 + ko);
    }
    async_wait0();
    __syncthreads();

    // scores: S = Q K^T  (16 q-rows x 32 keys per wave)
    // hoist all 4 K fragments, then 4 back-to-back WMMAs
    Frag kf[2][2];
#pragma unroll
    for (int j = 0; j < 2; ++j)
#pragma unroll
      for (int ks = 0; ks < 2; ++ks)
        load_frag(kf[j][ks], &Ks[(j * 16 + l16) * LDK + ks * 32 + half * 8]);
    v8f sfr[2];
#pragma unroll
    for (int j = 0; j < 2; ++j) {
      v8f a = {};
#pragma unroll
      for (int ks = 0; ks < 2; ++ks)
        a = __builtin_amdgcn_wmma_f32_16x16x32_bf16(
            false, qf[ks].v, false, kf[j][ks].v, (short)0, a, false, false);
      sfr[j] = a;
    }

    // online softmax (C layout: component p -> row p + 8*half; columns spread
    // over the 16-lane half -> shfl_xor masks 1..8 stay within the half)
#pragma unroll
    for (int p = 0; p < 8; ++p) {
      const int rq = q0 + wave * 16 + p + half * 8;
      float s0 = sfr[0][p] * 0.125f;       // 1/sqrt(64)
      float s1 = sfr[1][p] * 0.125f;
      if (k0 + l16 > rq)      s0 = -3.0e38f;
      if (k0 + 16 + l16 > rq) s1 = -3.0e38f;
      float mx = fmaxf(s0, s1);
#pragma unroll
      for (int m = 1; m < 16; m <<= 1) mx = fmaxf(mx, __shfl_xor(mx, m, 32));
      float mnew  = fmaxf(mrow[p], mx);
      float alpha = __expf(mrow[p] - mnew);
      float p0 = __expf(s0 - mnew);
      float p1 = __expf(s1 - mnew);
      float rs = p0 + p1;
#pragma unroll
      for (int m = 1; m < 16; m <<= 1) rs += __shfl_xor(rs, m, 32);
      lrow[p] = lrow[p] * alpha + rs;
      mrow[p] = mnew;
#pragma unroll
      for (int f = 0; f < 4; ++f) Oacc[f][p] *= alpha;
      bf16* pr = &Ps[(wave * 16 + p + half * 8) * LDP];
      pr[l16]      = (bf16)p0;
      pr[16 + l16] = (bf16)p1;
    }

    // O += P @ V   (per-wave P region; same-wave LDS ops are in order)
    // hoist P + all 4 V fragments, then 4 back-to-back WMMAs
    Frag pf, vfr[4];
    load_frag(pf, &Ps[(wave * 16 + l16) * LDP + half * 8]);
#pragma unroll
    for (int f = 0; f < 4; ++f)
      load_frag(vfr[f], &Vts[(f * 16 + l16) * LDV + half * 8]);
#pragma unroll
    for (int f = 0; f < 4; ++f)
      Oacc[f] = __builtin_amdgcn_wmma_f32_16x16x32_bf16(
          false, pf.v, false, vfr[f].v, (short)0, Oacc[f], false, false);
  }

  // normalize and write [B,S,NH*HD]
#pragma unroll
  for (int p = 0; p < 8; ++p) {
    const int s   = q0 + wave * 16 + p + half * 8;
    const float inv = 1.0f / lrow[p];
#pragma unroll
    for (int f = 0; f < 4; ++f)
      O[(size_t)(b * SS + s) * (NH * HD) + h * HD + f * 16 + l16] =
          (bf16)(Oacc[f][p] * inv);
  }
}

// ---------------------------------------------------------------------------
// SwiGLU activation: act = silu(gate) * up   (bf16 in/out, x4 vectorized)
// ---------------------------------------------------------------------------
__global__ void silu_mul_kernel(const bf16* __restrict__ g, const bf16* __restrict__ u,
                                bf16* __restrict__ out, int n4) {
  int i = blockIdx.x * blockDim.x + threadIdx.x;
  if (i < n4) {
    v4bf gv4 = reinterpret_cast<const v4bf*>(g)[i];
    v4bf uv4 = reinterpret_cast<const v4bf*>(u)[i];
    v4bf o;
#pragma unroll
    for (int j = 0; j < 4; ++j) {
      float gv = (float)gv4[j], uv = (float)uv4[j];
      o[j] = (bf16)(gv / (1.0f + __expf(-gv)) * uv);
    }
    reinterpret_cast<v4bf*>(out)[i] = o;
  }
}

// ---------------------------------------------------------------------------
// Host orchestration
// ---------------------------------------------------------------------------
extern "C" void kernel_launch(void* const* d_in, const int* in_sizes, int n_in,
                              void* d_out, int out_size, void* d_ws, size_t ws_size,
                              hipStream_t stream) {
  const float* hs    = (const float*)d_in[0];
  const float* wq    = (const float*)d_in[1];
  const float* wk    = (const float*)d_in[2];
  const float* wv    = (const float*)d_in[3];
  const float* wo    = (const float*)d_in[4];
  const float* wgate = (const float*)d_in[5];
  const float* wup   = (const float*)d_in[6];
  const float* wdown = (const float*)d_in[7];
  const float* n1w   = (const float*)d_in[8];
  const float* n2w   = (const float*)d_in[9];
  float* out = (float*)d_out;

  char* ws = (char*)d_ws;
  size_t off = 0;
  auto carve = [&](size_t bytes) -> void* {
    void* p = ws + off;
    off = (off + bytes + 255) & ~(size_t)255;
    return p;
  };

  bf16*  wq_t   = (bf16*)carve((size_t)HID * (NH * HD) * 2);   // [N,K] bf16
  bf16*  wk_t   = (bf16*)carve((size_t)HID * (NKV * HD) * 2);
  bf16*  wv_t   = (bf16*)carve((size_t)HID * (NKV * HD) * 2);
  bf16*  wo_t   = (bf16*)carve((size_t)(NH * HD) * HID * 2);
  bf16*  wg_t   = (bf16*)carve((size_t)HID * FF * 2);
  bf16*  wu_t   = (bf16*)carve((size_t)HID * FF * 2);
  bf16*  wd_t   = (bf16*)carve((size_t)FF * HID * 2);
  bf16*  xn_b   = (bf16*)carve((size_t)TOK * HID * 2);
  float* q_f    = (float*)carve((size_t)TOK * (NH * HD) * 4);
  float* k_f    = (float*)carve((size_t)TOK * (NKV * HD) * 4);
  float* v_f    = (float*)carve((size_t)TOK * (NKV * HD) * 4);
  bf16*  q_b    = (bf16*)carve((size_t)TOK * (NH * HD) * 2);
  bf16*  k_b    = (bf16*)carve((size_t)TOK * (NKV * HD) * 2);
  bf16*  vt_b   = (bf16*)carve((size_t)TOK * (NKV * HD) * 2);
  bf16*  attn_b = (bf16*)carve((size_t)TOK * (NH * HD) * 2);
  float* hid_f  = (float*)carve((size_t)TOK * HID * 4);
  bf16*  yn_b   = (bf16*)carve((size_t)TOK * HID * 2);
  bf16*  gate_b = (bf16*)carve((size_t)TOK * FF * 2);
  bf16*  up_b   = (bf16*)carve((size_t)TOK * FF * 2);
  bf16*  act_b  = (bf16*)carve((size_t)TOK * FF * 2);
  (void)ws_size; (void)in_sizes; (void)n_in; (void)out_size;

  auto tr = [&](const float* w, bf16* wt, int K, int N) {
    transpose_bf16_kernel<<<dim3(N / 32, K / 32), 256, 0, stream>>>(w, wt, K, N);
  };

  // 1) weights -> transposed bf16 [N,K]
  tr(wq,    wq_t, HID, NH * HD);
  tr(wk,    wk_t, HID, NKV * HD);
  tr(wv,    wv_t, HID, NKV * HD);
  tr(wo,    wo_t, NH * HD, HID);
  tr(wgate, wg_t, HID, FF);
  tr(wup,   wu_t, HID, FF);
  tr(wdown, wd_t, FF, HID);

  // 2) pre-attention RMSNorm
  rmsnorm_bf16_kernel<<<TOK, 256, 0, stream>>>(hs, n1w, xn_b);

  // 3) Q/K/V projections
  gemm_bf16_kernel<0, 0><<<dim3((NH * HD) / 128, TOK / 128), 256, 0, stream>>>(
      xn_b, wq_t, q_f, nullptr, nullptr, TOK, NH * HD, HID);
  gemm_bf16_kernel<0, 0><<<dim3((NKV * HD) / 128, TOK / 128), 256, 0, stream>>>(
      xn_b, wk_t, k_f, nullptr, nullptr, TOK, NKV * HD, HID);
  gemm_bf16_kernel<0, 0><<<dim3((NKV * HD) / 128, TOK / 128), 256, 0, stream>>>(
      xn_b, wv_t, v_f, nullptr, nullptr, TOK, NKV * HD, HID);

  // 4) RoPE (+ relayout); V transposed to [B,NKV,HD,S]
  rope_relayout_kernel<<<(BB * NH  * SS * 32) / 256, 256, 0, stream>>>(q_f, q_b, NH, 1, 0);
  rope_relayout_kernel<<<(BB * NKV * SS * 32) / 256, 256, 0, stream>>>(k_f, k_b, NKV, 1, 0);
  rope_relayout_kernel<<<(BB * NKV * SS * 32) / 256, 256, 0, stream>>>(v_f, vt_b, NKV, 0, 1);

  // 5) causal GQA flash attention
  attn_kernel<<<dim3(SS / 64, NH, BB), 128, 0, stream>>>(q_b, k_b, vt_b, attn_b);

  // 6) output projection + residual
  gemm_bf16_kernel<1, 0><<<dim3(HID / 128, TOK / 128), 256, 0, stream>>>(
      attn_b, wo_t, hid_f, nullptr, hs, TOK, HID, NH * HD);

  // 7) pre-MLP RMSNorm
  rmsnorm_bf16_kernel<<<TOK, 256, 0, stream>>>(hid_f, n2w, yn_b);

  // 8) gate / up projections (bf16 out)
  gemm_bf16_kernel<0, 1><<<dim3(FF / 128, TOK / 128), 256, 0, stream>>>(
      yn_b, wg_t, nullptr, gate_b, nullptr, TOK, FF, HID);
  gemm_bf16_kernel<0, 1><<<dim3(FF / 128, TOK / 128), 256, 0, stream>>>(
      yn_b, wu_t, nullptr, up_b, nullptr, TOK, FF, HID);

  // 9) SwiGLU
  silu_mul_kernel<<<((TOK * FF) / 4 + 255) / 256, 256, 0, stream>>>(
      gate_b, up_b, act_b, (TOK * FF) / 4);

  // 10) down projection + residual -> final output
  gemm_bf16_kernel<1, 0><<<dim3(HID / 128, TOK / 128), 256, 0, stream>>>(
      act_b, wd_t, out, nullptr, hid_f, TOK, HID, FF);
}
